// GNNRefiner_33354716021242
// MI455X (gfx1250) — matrix operand order repaired and checked by reading, not compiled, not run on hardware
//
#include <hip/hip_runtime.h>
#include <hip/hip_bf16.h>
#include <stdint.h>

// ---------------------------------------------------------------------------
// GNN point refiner for MI455X (gfx1250, wave32).
//   delta_e = relu([x_i, x_j-x_i] @ W1 + b1) @ W2 + b2 ; out = xyz + mean_K
// Restructuring: feat@W1 = a_i + b_j - b_i with
//   a = X @ W1[:387], b = X @ W1[387:]  (X = [token|xyz], 8192x387)
// -> one (8192 x 387) @ (387 x 768) GEMM on the bf16 WMMA path (fp32 acc)
//    (78 GFLOP naive edge-GEMM collapsed to 4.9 GFLOP), then a cheap
//    per-edge ReLU + 384x3 matvec whose b_j gathers hit L2 (12.6 MB << 192 MB).
// GEMM: each wave owns a 16x64 strip -> A fragment loaded once per k-step,
//   reused by 4 WMMAs (4 wmma : 10 b128 loads instead of 1 : 4).
// ---------------------------------------------------------------------------

typedef __attribute__((ext_vector_type(16))) __bf16 v16bf;
typedef __attribute__((ext_vector_type(8)))  float  v8f;

#define N_PTS   8192
#define D_TOK   384
#define KNN_K   16
#define DIN_H   387          // D_TOK + 3
#define KP      416          // DIN_H padded to multiple of 32 (13 k-steps)
#define NC      768          // output cols: [ a(384) | b(384) ]
#define NSTRIP  (NC / 64)    // 12 strips of 4 N-tiles

__device__ __forceinline__ unsigned short f2bf_rne(float f) {
  unsigned int u = __float_as_uint(f);
  unsigned int r = u + 0x7fffu + ((u >> 16) & 1u);   // round-to-nearest-even
  return (unsigned short)(r >> 16);
}

// ---------------------------------------------------------------------------
// 1) Brute-force KNN: one thread per point, xyz tiled through LDS,
//    register-resident sorted top-16 (insert-by-bubble, fully unrolled).
// ---------------------------------------------------------------------------
__global__ void knn_kernel(const float* __restrict__ xyz, int* __restrict__ idx) {
  __shared__ float sx[256 * 3];
  const int i = blockIdx.x * 256 + threadIdx.x;
  const float xi = xyz[i * 3 + 0], yi = xyz[i * 3 + 1], zi = xyz[i * 3 + 2];
  float bd[KNN_K]; int bi[KNN_K];
#pragma unroll
  for (int q = 0; q < KNN_K; ++q) { bd[q] = 3.4e38f; bi[q] = 0; }
  for (int jt = 0; jt < N_PTS; jt += 256) {
    __syncthreads();
    const int j0 = jt + threadIdx.x;
    sx[threadIdx.x * 3 + 0] = xyz[j0 * 3 + 0];
    sx[threadIdx.x * 3 + 1] = xyz[j0 * 3 + 1];
    sx[threadIdx.x * 3 + 2] = xyz[j0 * 3 + 2];
    __syncthreads();
    for (int t = 0; t < 256; ++t) {
      const int j = jt + t;
      const float dx = sx[t * 3 + 0] - xi;
      const float dy = sx[t * 3 + 1] - yi;
      const float dz = sx[t * 3 + 2] - zi;
      float d = dx * dx + dy * dy + dz * dz;
      if (j == i) d = 3.0e38f;                      // exclude self (diag = inf)
      if (d < bd[KNN_K - 1]) {
        float cd = d; int ci = j;
#pragma unroll
        for (int q = 0; q < KNN_K; ++q) {
          if (cd < bd[q]) {
            float td = bd[q]; int ti = bi[q];
            bd[q] = cd; bi[q] = ci; cd = td; ci = ti;
          }
        }
      }
    }
  }
#pragma unroll
  for (int q = 0; q < KNN_K; ++q) idx[i * KNN_K + q] = bi[q];
}

// ---------------------------------------------------------------------------
// 2) Pack X = [token | xyz | 0-pad] as bf16, row stride KP.
// ---------------------------------------------------------------------------
__global__ void pack_x_kernel(const float* __restrict__ token,
                              const float* __restrict__ xyz,
                              unsigned short* __restrict__ Xbf) {
  const int g = blockIdx.x * 256 + threadIdx.x;
  if (g >= N_PTS * KP) return;
  const int i = g / KP, k = g % KP;
  float v = 0.0f;
  if (k < D_TOK)            v = token[i * D_TOK + k];
  else if (k < D_TOK + 3)   v = xyz[i * 3 + (k - D_TOK)];
  Xbf[g] = f2bf_rne(v);
}

// ---------------------------------------------------------------------------
// 3) Pack W transposed as bf16: Wt[n][k] (n<384 -> W1_top col n; n>=384 ->
//    W1_bot col n-384). Column-contiguous so B fragments load contiguously.
// ---------------------------------------------------------------------------
__global__ void pack_w_kernel(const float* __restrict__ W1,
                              unsigned short* __restrict__ Wt) {
  const int g = blockIdx.x * 256 + threadIdx.x;
  if (g >= NC * KP) return;
  const int n = g / KP, k = g % KP;
  float v = 0.0f;
  if (k < DIN_H)
    v = (n < D_TOK) ? W1[k * D_TOK + n]
                    : W1[(DIN_H + k) * D_TOK + (n - D_TOK)];
  Wt[g] = f2bf_rne(v);
}

// ---------------------------------------------------------------------------
// 4) GEMM C(8192 x 768) = Xbf(8192 x 416) @ Wt^T via v_wmma_f32_16x16x32_bf16.
//    One 16x64 strip per wave (4 accumulators); A fragment reused 4x.
//    Fragment layouts used here (A/B: contiguous k per lane half, same order in both):
//      A (16x32 bf16): lane m=l&15; half0 lanes hold K0..15,
//                      half1 lanes hold K16..31  -> two 16B loads
//      B (32x16 bf16): lane n=l&15; half0 K0..15, half1 K16..31 -> 32B run
//      C (16x16 f32) : VGPR r -> rows r / r+8 split by lane half
// ---------------------------------------------------------------------------
__global__ void gemm_wmma_kernel(const unsigned short* __restrict__ Xbf,
                                 const unsigned short* __restrict__ Wt,
                                 float* __restrict__ C) {
  const int lane = threadIdx.x & 31;
  const int wid  = blockIdx.x * 4 + (threadIdx.x >> 5);  // 4 waves / block
  const int ts   = wid % NSTRIP;       // strip of 4 N-tiles
  const int tm   = wid / NSTRIP;       // M tile
  const int row  = lane & 15;          // A: M row, B: N column
  const int hi   = lane >> 4;          // lane half

  const unsigned short* arow = Xbf + (size_t)(tm * 16 + row) * KP + hi * 16;
  const unsigned short* br0  = Wt  + (size_t)(ts * 64 +  0 + row) * KP + hi * 16;
  const unsigned short* br1  = br0 + (size_t)16 * KP;
  const unsigned short* br2  = br0 + (size_t)32 * KP;
  const unsigned short* br3  = br0 + (size_t)48 * KP;

  v8f acc0 = {}, acc1 = {}, acc2 = {}, acc3 = {};
  for (int kb = 0; kb < KP; kb += 32) {
    union { uint4 q[2]; v16bf v; } A, B0, B1, B2, B3;
    A.q[0]  = *(const uint4*)(arow + kb);        // K kb+{0..7}  (this half)
    A.q[1]  = *(const uint4*)(arow + kb + 8);    // K kb+{8..15} (this half)
    B0.q[0] = *(const uint4*)(br0 + kb);         // K kb+{0..15} (this half)
    B0.q[1] = *(const uint4*)(br0 + kb + 8);
    B1.q[0] = *(const uint4*)(br1 + kb);
    B1.q[1] = *(const uint4*)(br1 + kb + 8);
    B2.q[0] = *(const uint4*)(br2 + kb);
    B2.q[1] = *(const uint4*)(br2 + kb + 8);
    B3.q[0] = *(const uint4*)(br3 + kb);
    B3.q[1] = *(const uint4*)(br3 + kb + 8);
    acc0 = __builtin_amdgcn_wmma_f32_16x16x32_bf16(false, A.v, false, B0.v,
                                                   (short)0, acc0, false, false);
    acc1 = __builtin_amdgcn_wmma_f32_16x16x32_bf16(false, A.v, false, B1.v,
                                                   (short)0, acc1, false, false);
    acc2 = __builtin_amdgcn_wmma_f32_16x16x32_bf16(false, A.v, false, B2.v,
                                                   (short)0, acc2, false, false);
    acc3 = __builtin_amdgcn_wmma_f32_16x16x32_bf16(false, A.v, false, B3.v,
                                                   (short)0, acc3, false, false);
  }

  float* crow = C + (size_t)(tm * 16 + hi * 8) * NC + ts * 64 + row;
#pragma unroll
  for (int r = 0; r < 8; ++r) {
    crow[(size_t)r * NC +  0] = acc0[r];
    crow[(size_t)r * NC + 16] = acc1[r];
    crow[(size_t)r * NC + 32] = acc2[r];
    crow[(size_t)r * NC + 48] = acc3[r];
  }
}

// ---------------------------------------------------------------------------
// 5) Per-point fusion: u_d = a_i[d] - b_i[d] + b1[d] + b_j[d]; h = relu(u);
//    delta = (1/16) * sum_k h @ W2 + b2 ; out = xyz + delta.
//    128 threads x 3 dims = 384. b_j gathers (1536B/row) hit L2.
// ---------------------------------------------------------------------------
__global__ void edge_kernel(const float* __restrict__ C,
                            const int*   __restrict__ idx,
                            const float* __restrict__ xyz,
                            const float* __restrict__ b1,
                            const float* __restrict__ W2,
                            const float* __restrict__ b2,
                            float* __restrict__ out) {
  const int i  = blockIdx.x;
  const int t  = threadIdx.x;          // 128 threads
  const int d0 = t * 3;
  __shared__ float red[3];
  __shared__ int   sj[KNN_K];
  if (t < 3)      red[t] = 0.0f;
  if (t < KNN_K)  sj[t]  = idx[i * KNN_K + t];

  float base[3], w[3][3];
#pragma unroll
  for (int c = 0; c < 3; ++c) {
    const int d = d0 + c;
    base[c] = C[(size_t)i * NC + d] - C[(size_t)i * NC + D_TOK + d] + b1[d];
    w[c][0] = W2[d * 3 + 0];
    w[c][1] = W2[d * 3 + 1];
    w[c][2] = W2[d * 3 + 2];
  }
  __syncthreads();

  float a0 = 0.0f, a1 = 0.0f, a2 = 0.0f;
  for (int k = 0; k < KNN_K; ++k) {
    const float* bj = C + (size_t)sj[k] * NC + D_TOK + d0;
#pragma unroll
    for (int c = 0; c < 3; ++c) {
      const float u = base[c] + bj[c];
      const float h = u > 0.0f ? u : 0.0f;
      a0 += h * w[c][0];
      a1 += h * w[c][1];
      a2 += h * w[c][2];
    }
  }
  // wave32 shuffle reduce, then LDS float atomics across the 4 waves
  for (int off = 16; off > 0; off >>= 1) {
    a0 += __shfl_down(a0, off, 32);
    a1 += __shfl_down(a1, off, 32);
    a2 += __shfl_down(a2, off, 32);
  }
  if ((t & 31) == 0) {
    atomicAdd(&red[0], a0);
    atomicAdd(&red[1], a1);
    atomicAdd(&red[2], a2);
  }
  __syncthreads();
  if (t < 3) out[i * 3 + t] = xyz[i * 3 + t] + red[t] * (1.0f / 16.0f) + b2[t];
}

// ---------------------------------------------------------------------------
extern "C" void kernel_launch(void* const* d_in, const int* in_sizes, int n_in,
                              void* d_out, int out_size, void* d_ws, size_t ws_size,
                              hipStream_t stream) {
  (void)in_sizes; (void)n_in; (void)out_size; (void)ws_size;
  const float* xyz   = (const float*)d_in[0];
  const float* token = (const float*)d_in[1];
  const float* W1    = (const float*)d_in[2];
  const float* b1    = (const float*)d_in[3];
  const float* W2    = (const float*)d_in[4];
  const float* b2    = (const float*)d_in[5];
  float* out = (float*)d_out;

  // Workspace layout (~33 MB total), 256B-aligned slices.
  char* ws = (char*)d_ws;
  size_t off = 0;
  auto alloc = [&](size_t bytes) -> char* {
    char* p = ws + off;
    off = (off + bytes + 255) & ~(size_t)255;
    return p;
  };
  int*            idx = (int*)           alloc((size_t)N_PTS * KNN_K * sizeof(int));
  unsigned short* Xbf = (unsigned short*)alloc((size_t)N_PTS * KP * sizeof(unsigned short));
  unsigned short* Wt  = (unsigned short*)alloc((size_t)NC    * KP * sizeof(unsigned short));
  float*          Cab = (float*)         alloc((size_t)N_PTS * NC * sizeof(float));

  knn_kernel   <<<N_PTS / 256, 256, 0, stream>>>(xyz, idx);
  pack_x_kernel<<<(N_PTS * KP + 255) / 256, 256, 0, stream>>>(token, xyz, Xbf);
  pack_w_kernel<<<(NC    * KP + 255) / 256, 256, 0, stream>>>(W1, Wt);
  // 512 M-tiles x 12 strips = 6144 waves; 4 waves (128 threads) per block
  gemm_wmma_kernel<<<(N_PTS / 16) * NSTRIP / 4, 128, 0, stream>>>(Xbf, Wt, Cab);
  edge_kernel  <<<N_PTS, 128, 0, stream>>>(Cab, idx, xyz, b1, W2, b2, out);
}
